// WordSentenceIntegrateBlock_70239895159359
// MI455X (gfx1250) — matrix operand order repaired and checked by reading, not compiled, not run on hardware
//
#include <hip/hip_runtime.h>

typedef __attribute__((ext_vector_type(16))) __bf16 v16bf;
typedef __attribute__((ext_vector_type(4)))  __bf16 v4bf;
typedef __attribute__((ext_vector_type(8)))  float  v8f;

namespace {
constexpr int kD  = 1024;         // sent_dim == N
constexpr int kK  = 2048;         // 2*D == K
constexpr int kS  = 255;
constexpr int TILE_M = 128;
constexpr int TILE_N = 128;
constexpr int KT     = 64;        // K staged per iteration (two wmma k-steps)
constexpr int NCHUNK = kK / KT;   // 32 K-chunks
constexpr int BUFE   = 8192;      // bf16 elements per (tensor,half) buffer
}

// ---- fragment-order packing (contiguous k order, same for A and B) ----
// A frag (16x32 bf16): lanes 0-15 hold M rows with K 0..15 packed 2/VGPR,
// lanes 16-31 hold K 16..31.
__device__ __forceinline__ int packA(int r, int k) {
  const int ms = r >> 4, lm = r & 15;
  const int ks = k >> 5, kk = k & 31;
  const int ln = lm + (((kk >> 4) & 1) << 4);
  const int i  = kk & 15;
  return (((ms << 1) + ks) * 32 + ln) * 16 + i;   // blob*512 + lane*16 + elem
}
// B frag (32x16 bf16): lanes 0-15 hold N cols with K 0..15 packed 2/VGPR,
// lanes 16-31 hold K 16..31.
__device__ __forceinline__ int packB(int k, int n) {
  const int ns = n >> 4, lnn = n & 15;
  const int ks = k >> 5, kk = k & 31;
  const int ln = lnn + (((kk >> 4) & 1) << 4);
  const int i  = kk & 15;
  return (((ns << 1) + ks) * 32 + ln) * 16 + i;
}

// fp32 -> bf16 hi/lo split (x ~= hi + lo; dropped lo*lo term ~2^-16 relative)
__device__ __forceinline__ void split4(float4 v, v4bf& hi, v4bf& lo) {
  const __bf16 h0 = (__bf16)v.x, h1 = (__bf16)v.y, h2 = (__bf16)v.z, h3 = (__bf16)v.w;
  hi[0] = h0; hi[1] = h1; hi[2] = h2; hi[3] = h3;
  lo[0] = (__bf16)(v.x - (float)h0);
  lo[1] = (__bf16)(v.y - (float)h1);
  lo[2] = (__bf16)(v.z - (float)h2);
  lo[3] = (__bf16)(v.w - (float)h3);
}

__device__ __forceinline__ v8f wmma_bf16(v16bf a, v16bf b, v8f c) {
  // 8 args: (neg_a, A, neg_b, B, c_mod, C, reuse_a, reuse_b)
  return __builtin_amdgcn_wmma_f32_16x16x32_bf16(false, a, false, b, (short)0, c,
                                                 false, false);
}

// ---- one-shot W pre-pack: f32 [N,K] -> hi/lo bf16 in frag-blob order ----
// ws layout: region (nt, kc) = 16384 bf16 (16 KiB hi @ +0, 16 KiB lo @ +8192)
__global__ __launch_bounds__(256)
void prepack_w(const float* __restrict__ W, __bf16* __restrict__ Bpk) {
  const int nt  = blockIdx.x;     // 0..7  (128-col band)
  const int kc  = blockIdx.y;     // 0..31 (64-deep K chunk)
  const int tid = threadIdx.x;
  __bf16* dst = Bpk + (size_t)(nt * NCHUNK + kc) * (2 * BUFE);
  const int k0 = kc * KT;
  #pragma unroll
  for (int t = 0; t < 8; ++t) {
    const int f  = tid + t * 256;          // 0..2047
    const int nl = f >> 4;                 // 0..127
    const int k4 = (f & 15) << 2;          // 0..60 step 4 (coalesced along K)
    const float4 wv =
        *(const float4*)(W + (size_t)(nt * TILE_N + nl) * kK + (k0 + k4));
    v4bf bh, bl;
    split4(wv, bh, bl);
    const int off = packB(k4, nl);
    *(v4bf*)(dst + off)        = bh;
    *(v4bf*)(dst + BUFE + off) = bl;
  }
}

// ---- staging helpers (write one LDS buffer) ----
__device__ __forceinline__ void stageA(const float* __restrict__ words,
                                       const float* __restrict__ sents,
                                       const int*   __restrict__ smap,
                                       int mBase, int k0, int tid,
                                       __bf16* dAh, __bf16* dAl) {
  const bool wordsRegion = (k0 < kD);      // KT-aligned split at K=1024
  #pragma unroll
  for (int t = 0; t < 8; ++t) {
    const int f  = tid + t * 256;
    const int r  = f >> 4;                 // 0..127
    const int k4 = (f & 15) << 2;
    const int m  = mBase + r;
    float4 av;
    if (wordsRegion) {
      av = *(const float4*)(words + (size_t)m * kD + (k0 + k4));
    } else {
      const int s = smap[m];
      if (s >= 0) {
        const int b = m >> 12;             // m / L
        av = *(const float4*)(sents + ((size_t)b * kS + s) * kD + (k0 - kD + k4));
      } else {
        av = make_float4(0.f, 0.f, 0.f, 0.f);
      }
    }
    v4bf ah, al;
    split4(av, ah, al);
    const int off = packA(r, k4);
    *(v4bf*)(dAh + off) = ah;
    *(v4bf*)(dAl + off) = al;
  }
}

// Pre-packed B staging via the gfx1250 async LDS-DMA path: 16B per lane per
// op, tracked on ASYNCcnt (no VGPR round-trip, no ds_store issue cost).
// LDS byte address = low 32 bits of the generic pointer (aperture rule,
// ISA 10.2).
__device__ __forceinline__ void async_cp16(const void* gsrc, void* ldst) {
  const unsigned long long ga = (unsigned long long)(uintptr_t)gsrc;
  const unsigned int       la = (unsigned int)(uintptr_t)ldst;
  asm volatile("global_load_async_to_lds_b128 %0, %1, off"
               :: "v"(la), "v"(ga) : "memory");
}

__device__ __forceinline__ void stageB_async(const __bf16* __restrict__ src,
                                             int tid, __bf16* dBh, __bf16* dBl) {
  const char* s  = (const char*)src;        // 16 KiB hi then 16 KiB lo
  char* dh = (char*)dBh;
  char* dl = (char*)dBl;
  #pragma unroll
  for (int t = 0; t < 4; ++t) {
    const size_t o = (size_t)(tid + t * 256) * 16;
    async_cp16(s + o, dh + o);
  }
  #pragma unroll
  for (int t = 0; t < 4; ++t) {
    const size_t o = (size_t)(tid + t * 256) * 16;
    async_cp16(s + 16384 + o, dl + o);
  }
}

__device__ __forceinline__ void stageB_convert(const float* __restrict__ W,
                                               int nBase, int k0, int tid,
                                               __bf16* dBh, __bf16* dBl) {
  #pragma unroll
  for (int t = 0; t < 8; ++t) {
    const int f  = tid + t * 256;
    const int nl = f >> 4;
    const int k4 = (f & 15) << 2;
    const float4 wv = *(const float4*)(W + (size_t)(nBase + nl) * kK + (k0 + k4));
    v4bf bh, bl;
    split4(wv, bh, bl);
    const int off = packB(k4, nl);
    *(v4bf*)(dBh + off) = bh;
    *(v4bf*)(dBl + off) = bl;
  }
}

// ---- main fused gather+GEMM+bias+ReLU, double-buffered LDS ----
template <bool PREPACKED>
__global__ __launch_bounds__(256)
void wsint_fused_gemm_wmma(const float* __restrict__ words,   // [B,L,D]
                           const float* __restrict__ sents,   // [B,S,D]
                           const float* __restrict__ W,       // [D, 2D]
                           const float* __restrict__ bias,    // [D]
                           const int*   __restrict__ smap,    // [B,L]
                           const __bf16* __restrict__ Bpk,    // packed W or null
                           float*       __restrict__ out) {   // [B,L,D]
  extern __shared__ __align__(32) __bf16 smem[];   // 128 KiB
  __bf16* sAh = smem;                  // [2][BUFE]
  __bf16* sAl = smem + 2 * BUFE;
  __bf16* sBh = smem + 4 * BUFE;
  __bf16* sBl = smem + 6 * BUFE;

  const int tid   = threadIdx.x;
  const int lane  = tid & 31;
  const int wave  = tid >> 5;          // 8 waves
  const int waveM = wave >> 1;         // 0..3 -> 32-row band
  const int waveN = wave & 1;          // 0..1 -> 64-col band
  const int nt    = blockIdx.x;
  const int mBase = blockIdx.y * TILE_M;
  const int nBase = nt * TILE_N;

  v8f acc[2][4] = {};                  // [mSub][nSub] 16x16 f32 accumulators

  auto stage = [&](int kc, int buf) {
    stageA(words, sents, smap, mBase, kc * KT, tid,
           sAh + buf * BUFE, sAl + buf * BUFE);
    if constexpr (PREPACKED) {
      stageB_async(Bpk + (size_t)(nt * NCHUNK + kc) * (2 * BUFE), tid,
                   sBh + buf * BUFE, sBl + buf * BUFE);
    } else {
      stageB_convert(W, nBase, kc * KT, tid,
                     sBh + buf * BUFE, sBl + buf * BUFE);
    }
  };

  stage(0, 0);                          // prologue
  for (int kc = 0; kc < NCHUNK; ++kc) {
    if constexpr (PREPACKED) {
      // inline-asm async ops are invisible to the compiler's counters;
      // drain our wave's B staging of buf[kc&1] before publishing it.
      asm volatile("s_wait_asynccnt 0x0" ::: "memory");
    }
    __syncthreads();                    // buf[kc&1] staging complete (all waves)
    if (kc + 1 < NCHUNK) stage(kc + 1, (kc + 1) & 1);  // overlap with compute

    const int cb = (kc & 1) * BUFE;
    #pragma unroll
    for (int ks = 0; ks < 2; ++ks) {
      v16bf aH[2], aL[2];
      #pragma unroll
      for (int ms = 0; ms < 2; ++ms) {
        const int blob = (((waveM * 2 + ms) << 1) + ks) * 512 + lane * 16;
        aH[ms] = *(const v16bf*)(sAh + cb + blob);
        aL[ms] = *(const v16bf*)(sAl + cb + blob);
      }
      #pragma unroll
      for (int ns = 0; ns < 4; ++ns) {
        const int blob = (((waveN * 4 + ns) << 1) + ks) * 512 + lane * 16;
        const v16bf bH = *(const v16bf*)(sBh + cb + blob);
        const v16bf bL = *(const v16bf*)(sBl + cb + blob);
        #pragma unroll
        for (int ms = 0; ms < 2; ++ms) {
          acc[ms][ns] = wmma_bf16(aH[ms], bH, acc[ms][ns]);  // hi*hi
          acc[ms][ns] = wmma_bf16(aH[ms], bL, acc[ms][ns]);  // hi*lo
          acc[ms][ns] = wmma_bf16(aL[ms], bH, acc[ms][ns]);  // lo*hi
        }
      }
    }
  }

  // ---- epilogue: bias + ReLU, C/D layout per ISA 7.12.2 ----
  const int laneN = lane & 15;
  const int laneH = lane >> 4;
  #pragma unroll
  for (int ms = 0; ms < 2; ++ms) {
    #pragma unroll
    for (int ns = 0; ns < 4; ++ns) {
      const int n  = nBase + waveN * 64 + ns * 16 + laneN;
      const float bn = bias[n];
      const int m0 = mBase + waveM * 32 + ms * 16 + laneH * 8;
      #pragma unroll
      for (int r = 0; r < 8; ++r) {
        const float v = acc[ms][ns][r] + bn;
        out[(size_t)(m0 + r) * kD + n] = v > 0.f ? v : 0.f;
      }
    }
  }
}

extern "C" void kernel_launch(void* const* d_in, const int* in_sizes, int n_in,
                              void* d_out, int out_size, void* d_ws, size_t ws_size,
                              hipStream_t stream) {
  (void)in_sizes; (void)n_in; (void)out_size;
  const float* words = (const float*)d_in[0];   // [8,4096,1024] f32
  const float* sents = (const float*)d_in[1];   // [8,255,1024]  f32
  const float* W     = (const float*)d_in[2];   // [1024,2048]   f32
  const float* bias  = (const float*)d_in[3];   // [1024]        f32
  const int*   smap  = (const int*)d_in[4];     // [8,4096]      i32
  float* out = (float*)d_out;                   // [8,4096,1024] f32

  const int M = 8 * 4096;
  dim3 grid(kD / TILE_N, M / TILE_M);           // (8, 256), N fastest for A reuse
  dim3 block(256);
  constexpr size_t LDS_BYTES = (size_t)8 * BUFE * sizeof(__bf16);      // 128 KiB
  const size_t needWs = (size_t)8 * NCHUNK * 2 * BUFE * sizeof(__bf16); // 8 MiB

  if (d_ws != nullptr && ws_size >= needWs) {
    __bf16* Bpk = (__bf16*)d_ws;
    prepack_w<<<dim3(8, NCHUNK), block, 0, stream>>>(W, Bpk);
    wsint_fused_gemm_wmma<true><<<grid, block, LDS_BYTES, stream>>>(
        words, sents, W, bias, smap, Bpk, out);
  } else {
    wsint_fused_gemm_wmma<false><<<grid, block, LDS_BYTES, stream>>>(
        words, sents, W, bias, smap, nullptr, out);
  }
}
